// MPNN_85426899517547
// MI455X (gfx1250) — matrix operand order repaired and checked by reading, not compiled, not run on hardware
//
#include <hip/hip_runtime.h>
#include <hip/hip_bf16.h>
#include <math.h>

// ---------------- problem constants (from reference) ----------------
#define NATOM  96
#define NNEI   95
#define NPAIR  (NATOM * NNEI)        // 9120
#define NWAVE_ 8
#define NC_    128
#define RMAXL_ 3
#define NSPH_  9
#define MPL_   2
#define NCG_   32
#define NCGU_  20
#define EMB_   8
#define DH_    64
#define DNEIGH (2 * NWAVE_ + RMAXL_ * NC_)   // 400
#define DDEN   (RMAXL_ * NC_)                // 384

typedef __attribute__((ext_vector_type(16))) __bf16 v16bf;
typedef __attribute__((ext_vector_type(8)))  float  v8f;

// ---------------- WMMA fragment helpers (CDNA5 ISA 7.12.2 layouts) ----------------
// A: 16x32 bf16.  lanes 0-15 -> M=lane, lanes 16-31 -> M=lane-16.
// element e (0..15): K = kbase + e + half*8 + (e>=8 ? 8 : 0)
// -> two contiguous 16B runs per lane => 2x ds_load_b128
__device__ __forceinline__ v16bf loadA_lds(const __bf16* x, int stride, int kbase, int lane) {
  const int half = lane >> 4, r = lane & 15;
  v16bf a;
#pragma unroll
  for (int e = 0; e < 16; ++e) {
    int k = kbase + e + half * 8 + ((e & 8) ? 8 : 0);
    a[e] = x[r * stride + k];
  }
  return a;
}
// B fragments are pre-packed (bf16, fragment-layout, zero-padded): lane reads 16
// contiguous bf16 (32 bytes) => 2x global_load_b128, no guards, no cvt.
__device__ __forceinline__ v16bf loadB_packed(const __bf16* __restrict__ P, int tile, int lane) {
  return *reinterpret_cast<const v16bf*>(P + (((size_t)tile * 32 + lane) << 4));
}

__device__ __forceinline__ float silu_f(float x) { return x / (1.0f + __expf(-x)); }

// ---------------- weight pre-pack: row-major f32 (Krows x ncols) -> WMMA B tiles ----------------
// tile (kt, ct): 32 lanes x 16 bf16; element e of lane -> K = kt*32 + e + half*8 + (e>=8 ? 8 : 0), N = ct*16 + (lane&15)
__global__ void pack_b(const float* __restrict__ W, int Krows, int ncols,
                       int ntiles, int total, __bf16* __restrict__ out) {
  int idx = blockIdx.x * blockDim.x + threadIdx.x;
  if (idx >= total) return;
  int e = idx & 15;
  int lane = (idx >> 4) & 31;
  int tile = idx >> 9;
  int ct = tile % ntiles, kt = tile / ntiles;
  int half = lane >> 4, r = lane & 15;
  int k = kt * 32 + e + half * 8 + ((e & 8) ? 8 : 0);
  int col = ct * 16 + r;
  float v = (k < Krows && col < ncols) ? W[(size_t)k * ncols + col] : 0.0f;
  out[idx] = (__bf16)v;
}

// ---------------- generic fused 3-layer MLP (bf16 WMMA, f32 accumulate) ----------------
// X: M x din (f32); packed weights; Y: M x dout (f32)
#define MLP_WAVES 2
#define KMAX 384
__global__ void __launch_bounds__(MLP_WAVES * 32)
mlp_wmma(const float* __restrict__ X, int M, int din,
         const __bf16* __restrict__ W0p, const float* __restrict__ b0,
         const __bf16* __restrict__ W1p, const float* __restrict__ b1,
         const __bf16* __restrict__ Wop, const float* __restrict__ bo,
         int dout, float* __restrict__ Y) {
  __shared__ __bf16 sx[MLP_WAVES][16][KMAX];      // 24 KB
  __shared__ __bf16 sh[MLP_WAVES][2][16][DH_];    //  8 KB
  const int tid = threadIdx.x;
  const int w = tid >> 5, lane = tid & 31;
  const int half = lane >> 4, r = lane & 15;
  const int row0 = (blockIdx.x * MLP_WAVES + w) * 16;
  const int kpad = ((din + 31) >> 5) << 5;
  const int kt0 = kpad >> 5;

  // stage X tile into LDS as bf16, zero padded to kpad
  for (int idx = lane; idx < 16 * kpad; idx += 32) {
    int rr = idx / kpad, kk = idx - rr * kpad;
    int gr = row0 + rr;
    float v = (gr < M && kk < din) ? X[(size_t)gr * din + kk] : 0.0f;
    sx[w][rr][kk] = (__bf16)v;
  }
  __syncthreads();

  // layer 1: (16 x kpad) @ (kpad x 64) + b0, silu
  for (int ct = 0; ct < DH_ / 16; ++ct) {
    v8f c = {};
    for (int kt = 0; kt < kt0; ++kt) {
      v16bf a = loadA_lds(&sx[w][0][0], KMAX, kt * 32, lane);
      v16bf b = loadB_packed(W0p, kt * (DH_ / 16) + ct, lane);
      c = __builtin_amdgcn_wmma_f32_16x16x32_bf16(false, a, false, b, (short)0, c, false, false);
    }
    float bb = b0[ct * 16 + r];
#pragma unroll
    for (int vv = 0; vv < 8; ++vv)
      sh[w][0][half * 8 + vv][ct * 16 + r] = (__bf16)silu_f(c[vv] + bb);
  }
  __syncthreads();

  // layer 2: (16 x 64) @ (64 x 64) + b1, silu
  for (int ct = 0; ct < DH_ / 16; ++ct) {
    v8f c = {};
#pragma unroll
    for (int kt = 0; kt < DH_ / 32; ++kt) {
      v16bf a = loadA_lds(&sh[w][0][0][0], DH_, kt * 32, lane);
      v16bf b = loadB_packed(W1p, kt * (DH_ / 16) + ct, lane);
      c = __builtin_amdgcn_wmma_f32_16x16x32_bf16(false, a, false, b, (short)0, c, false, false);
    }
    float bb = b1[ct * 16 + r];
#pragma unroll
    for (int vv = 0; vv < 8; ++vv)
      sh[w][1][half * 8 + vv][ct * 16 + r] = (__bf16)silu_f(c[vv] + bb);
  }
  __syncthreads();

  // layer 3: (16 x 64) @ (64 x dout) + bo
  const int nct = (dout + 15) >> 4;
  for (int ct = 0; ct < nct; ++ct) {
    v8f c = {};
#pragma unroll
    for (int kt = 0; kt < DH_ / 32; ++kt) {
      v16bf a = loadA_lds(&sh[w][1][0][0], DH_, kt * 32, lane);
      v16bf b = loadB_packed(Wop, kt * nct + ct, lane);
      c = __builtin_amdgcn_wmma_f32_16x16x32_bf16(false, a, false, b, (short)0, c, false, false);
    }
    int col = ct * 16 + r;
    float bb = (col < dout) ? bo[col] : 0.0f;
#pragma unroll
    for (int vv = 0; vv < 8; ++vv) {
      int gr = row0 + half * 8 + vv;
      if (gr < M && col < dout) Y[(size_t)gr * dout + col] = c[vv] + bb;
    }
  }
}

// ---------------- per-pair geometry: distvec, distance, real spherical harmonics ----------------
__global__ void pair_geom(const float* __restrict__ cart, const int* __restrict__ nl,
                          float* __restrict__ sph, float* __restrict__ dist) {
  int p = blockIdx.x * blockDim.x + threadIdx.x;
  if (p >= NPAIR) return;
  int i = nl[p], j = nl[NPAIR + p];
  float dx = cart[j * 3 + 0] - cart[i * 3 + 0];
  float dy = cart[j * 3 + 1] - cart[i * 3 + 1];
  float dz = cart[j * 3 + 2] - cart[i * 3 + 2];
  float d = sqrtf(dx * dx + dy * dy + dz * dz);
  float inv = 1.0f / d;
  float x = dx * inv, y = dy * inv, z = dz * inv;
  const float c0 = 0.28209479177387814f, c1 = 0.4886025119029199f;
  const float c2a = 1.0925484305920792f, c2b = 0.31539156525252005f, c2c = 0.5462742152960396f;
  float* s = sph + (size_t)p * NSPH_;
  s[0] = c0; s[1] = c1 * y; s[2] = c1 * z; s[3] = c1 * x;
  s[4] = c2a * x * y; s[5] = c2a * y * z; s[6] = c2b * (3.0f * z * z - 1.0f);
  s[7] = c2a * x * z; s[8] = c2c * (x * x - y * y);
  dist[p] = d;
}

// ---------------- radial basis: exp(-(w*(d-mu))^2) from emb_neigh cols [0:16) ----------------
__global__ void radial_k(const float* __restrict__ embN, const float* __restrict__ dist,
                         float* __restrict__ radial) {
  int p = blockIdx.x * blockDim.x + threadIdx.x;
  if (p >= NPAIR) return;
  float d = dist[p];
  const float* e = embN + (size_t)p * DNEIGH;
#pragma unroll
  for (int m = 0; m < NWAVE_; ++m) {
    float t = e[m] * (d - e[NWAVE_ + m]);
    radial[(size_t)p * NWAVE_ + m] = __expf(-t * t);
  }
}

// ---------------- initial center_orbital + density (block = atom, thread = channel j) ----------------
__global__ void __launch_bounds__(NC_)
center0_k(const float* __restrict__ cc, const float* __restrict__ radial,
          const float* __restrict__ sph, const float* __restrict__ embN,
          float* __restrict__ center, float* __restrict__ density) {
  __shared__ float scc[RMAXL_ * NC_ * NWAVE_];  // contracted_coeff[0] slice, 12 KB
  const int a = blockIdx.x, j = threadIdx.x;
  for (int idx = j; idx < RMAXL_ * NC_ * NWAVE_; idx += NC_) scc[idx] = cc[idx];
  __syncthreads();
  const int il[NSPH_] = {0, 1, 1, 1, 2, 2, 2, 2, 2};
  float acc[NSPH_];
#pragma unroll
  for (int k = 0; k < NSPH_; ++k) acc[k] = 0.0f;
  for (int n = 0; n < NNEI; ++n) {
    int p = a * NNEI + n;
    float r8[NWAVE_];
#pragma unroll
    for (int m = 0; m < NWAVE_; ++m) r8[m] = radial[(size_t)p * NWAVE_ + m];
    float wr[RMAXL_];
#pragma unroll
    for (int l = 0; l < RMAXL_; ++l) {
      float s = 0.0f;
#pragma unroll
      for (int m = 0; m < NWAVE_; ++m) s += scc[(l * NC_ + j) * NWAVE_ + m] * r8[m];
      wr[l] = s * embN[(size_t)p * DNEIGH + 2 * NWAVE_ + l * NC_ + j];
    }
#pragma unroll
    for (int k = 0; k < NSPH_; ++k) acc[k] += wr[il[k]] * sph[(size_t)p * NSPH_ + k];
  }
  const float inv = 1.0f / (float)NNEI;
  float dl[RMAXL_] = {0.0f, 0.0f, 0.0f};
#pragma unroll
  for (int k = 0; k < NSPH_; ++k) {
    float v = acc[k] * inv;
    center[((size_t)a * NSPH_ + k) * NC_ + j] = v;
    dl[il[k]] += v * v;
  }
#pragma unroll
  for (int l = 0; l < RMAXL_; ++l) density[((size_t)a * RMAXL_ + l) * NC_ + j] = dl[l];
}

// ---------------- message-passing interaction (block = atom, thread = channel j) ----------------
__global__ void __launch_bounds__(NC_)
interact_k(int t, const float* __restrict__ cc_all, const float* __restrict__ radial,
           const float* __restrict__ sph, const float* __restrict__ embC,
           const float* __restrict__ coeff, const int* __restrict__ nl,
           const int* __restrict__ i1, const int* __restrict__ i2,
           const int* __restrict__ iadd, const int* __restrict__ icg,
           const float* __restrict__ ens, const float* __restrict__ center_old,
           float* __restrict__ center_new, float* __restrict__ density) {
  __shared__ float scc[RMAXL_ * NC_ * NWAVE_];  // contracted_coeff[t+1] slice
  __shared__ float scg[NCG_];
  __shared__ int s1[NCG_], s2[NCG_], sa[NCG_], sic[NCG_];
  __shared__ float sens[NCG_];
  const int a = blockIdx.x, j = threadIdx.x;
  const float* ccb = cc_all + (size_t)(t + 1) * RMAXL_ * NC_ * NWAVE_;
  for (int idx = j; idx < RMAXL_ * NC_ * NWAVE_; idx += NC_) scc[idx] = ccb[idx];
  if (j < NCG_) { s1[j] = i1[j]; s2[j] = i2[j]; sa[j] = iadd[j]; sic[j] = icg[j]; sens[j] = ens[j]; }
  __syncthreads();
  const int il[NSPH_] = {0, 1, 1, 1, 2, 2, 2, 2, 2};
  float acc[NSPH_];
#pragma unroll
  for (int k = 0; k < NSPH_; ++k) acc[k] = 0.0f;
  for (int n = 0; n < NNEI; ++n) {
    int p = a * NNEI + n;
    int jn = nl[NPAIR + p];
    if (j < NCG_) scg[j] = embC[(size_t)p * (MPL_ * NCGU_) + t * NCGU_ + sic[j]] * sens[j];
    __syncthreads();
    float r8[NWAVE_];
#pragma unroll
    for (int m = 0; m < NWAVE_; ++m) r8[m] = radial[(size_t)p * NWAVE_ + m];
    float wr[RMAXL_];
#pragma unroll
    for (int l = 0; l < RMAXL_; ++l) {
      float s = 0.0f;
#pragma unroll
      for (int m = 0; m < NWAVE_; ++m) s += scc[(l * NC_ + j) * NWAVE_ + m] * r8[m];
      wr[l] = s * coeff[((size_t)jn * RMAXL_ + l) * NC_ + j];
    }
    float orb[NSPH_];
#pragma unroll
    for (int k = 0; k < NSPH_; ++k) orb[k] = wr[il[k]] * sph[(size_t)p * NSPH_ + k];
    const float* co = center_old + (size_t)jn * NSPH_ * NC_ + j;
#pragma unroll
    for (int c = 0; c < NCG_; ++c)
      acc[sa[c]] += co[s1[c] * NC_] * orb[s2[c]] * scg[c];
    __syncthreads();
  }
  const float inv = 1.0f / (float)NNEI;
  const float rs2 = 0.70710678118654752f;  // 1/sqrt(2)
  float dl[RMAXL_] = {0.0f, 0.0f, 0.0f};
#pragma unroll
  for (int k = 0; k < NSPH_; ++k) {
    float v = acc[k] * inv;
    center_new[((size_t)a * NSPH_ + k) * NC_ + j] = v;
    dl[il[k]] += v * v;
  }
#pragma unroll
  for (int l = 0; l < RMAXL_; ++l) {
    size_t o = ((size_t)a * RMAXL_ + l) * NC_ + j;
    density[o] = (density[o] + dl[l]) * rs2;
  }
}

// ---------------- final scalar reduction ----------------
__global__ void reduce_k(const float* __restrict__ y, float* __restrict__ out) {
  __shared__ float s[128];
  int t = threadIdx.x;
  s[t] = (t < NATOM) ? y[t] : 0.0f;
  __syncthreads();
  for (int off = 64; off > 0; off >>= 1) {
    if (t < off) s[t] += s[t + off];
    __syncthreads();
  }
  if (t == 0) out[0] = s[0];
}

// ---------------- host-side orchestration ----------------
static inline size_t packed_elems(int Krows, int ncols) {
  return (size_t)((Krows + 31) / 32) * ((ncols + 15) / 16) * 512;
}

extern "C" void kernel_launch(void* const* d_in, const int* in_sizes, int n_in,
                              void* d_out, int out_size, void* d_ws, size_t ws_size,
                              hipStream_t stream) {
  (void)in_sizes; (void)n_in; (void)out_size; (void)ws_size;
  // Top-level inputs in setup_inputs() insertion order; 'params' expands as a JAX
  // pytree (dict keys sorted): emb_cg, emb_neigh, mp[0], mp[1], out; each MLP is
  // {W0, W1, Wo, b0, b1, bo} (sorted).
  const float* cart        = (const float*)d_in[0];
  const float* input_embed = (const float*)d_in[1];
  const float* ens_cg      = (const float*)d_in[2];
  const float* ccoeff      = (const float*)d_in[3];
  const float* cgW0 = (const float*)d_in[4],  *cgW1 = (const float*)d_in[5],  *cgWo = (const float*)d_in[6];
  const float* cgb0 = (const float*)d_in[7],  *cgb1 = (const float*)d_in[8],  *cgbo = (const float*)d_in[9];
  const float* nW0  = (const float*)d_in[10], *nW1  = (const float*)d_in[11], *nWo  = (const float*)d_in[12];
  const float* nb0  = (const float*)d_in[13], *nb1  = (const float*)d_in[14], *nbo  = (const float*)d_in[15];
  const float* mW0[MPL_] = {(const float*)d_in[16], (const float*)d_in[22]};
  const float* mW1[MPL_] = {(const float*)d_in[17], (const float*)d_in[23]};
  const float* mWo[MPL_] = {(const float*)d_in[18], (const float*)d_in[24]};
  const float* mb0[MPL_] = {(const float*)d_in[19], (const float*)d_in[25]};
  const float* mb1[MPL_] = {(const float*)d_in[20], (const float*)d_in[26]};
  const float* mbo[MPL_] = {(const float*)d_in[21], (const float*)d_in[27]};
  const float* oW0 = (const float*)d_in[28], *oW1 = (const float*)d_in[29], *oWo = (const float*)d_in[30];
  const float* ob0 = (const float*)d_in[31], *ob1 = (const float*)d_in[32], *obo = (const float*)d_in[33];
  const int* nl   = (const int*)d_in[34];
  const int* i1   = (const int*)d_in[36];
  const int* i2   = (const int*)d_in[37];
  const int* iadd = (const int*)d_in[38];
  const int* icg  = (const int*)d_in[39];

  float* ws = (float*)d_ws;
  size_t o = 0;
  float* sph     = ws + o; o += (size_t)NPAIR * NSPH_;
  float* dist    = ws + o; o += (size_t)NPAIR;
  float* radial  = ws + o; o += (size_t)NPAIR * NWAVE_;
  float* embN    = ws + o; o += (size_t)NPAIR * DNEIGH;
  float* embC    = ws + o; o += (size_t)NPAIR * MPL_ * NCGU_;
  float* cenA    = ws + o; o += (size_t)NATOM * NSPH_ * NC_;
  float* cenB    = ws + o; o += (size_t)NATOM * NSPH_ * NC_;
  float* density = ws + o; o += (size_t)NATOM * DDEN;
  float* coeff   = ws + o; o += (size_t)NATOM * DDEN;
  float* yout    = ws + o; o += (size_t)NATOM;
  o = (o + 7) & ~(size_t)7;                 // 32B-align the bf16 region
  __bf16* wsb = (__bf16*)(ws + o);

  // carve packed-weight buffers (bf16)
  size_t ob = 0;
  auto carve = [&](int K, int Nn) { __bf16* p = wsb + ob; ob += packed_elems(K, Nn); return p; };
  __bf16 *nW0p = carve(EMB_, DH_), *nW1p = carve(DH_, DH_), *nWop = carve(DH_, DNEIGH);
  __bf16 *cgW0p = carve(EMB_, DH_), *cgW1p = carve(DH_, DH_), *cgWop = carve(DH_, MPL_ * NCGU_);
  __bf16 *mW0p[MPL_], *mW1p[MPL_], *mWop[MPL_];
  for (int t = 0; t < MPL_; ++t) {
    mW0p[t] = carve(DDEN, DH_); mW1p[t] = carve(DH_, DH_); mWop[t] = carve(DH_, DDEN);
  }
  __bf16 *oW0p = carve(DDEN, DH_), *oW1p = carve(DH_, DH_), *oWop = carve(DH_, 1);

  auto pack = [&](const float* W, int K, int Nn, __bf16* out) {
    int nt = (Nn + 15) / 16;
    int total = (int)packed_elems(K, Nn);
    pack_b<<<(total + 255) / 256, 256, 0, stream>>>(W, K, Nn, nt, total, out);
  };
  pack(nW0, EMB_, DH_, nW0p);   pack(nW1, DH_, DH_, nW1p);   pack(nWo, DH_, DNEIGH, nWop);
  pack(cgW0, EMB_, DH_, cgW0p); pack(cgW1, DH_, DH_, cgW1p); pack(cgWo, DH_, MPL_ * NCGU_, cgWop);
  for (int t = 0; t < MPL_; ++t) {
    pack(mW0[t], DDEN, DH_, mW0p[t]); pack(mW1[t], DH_, DH_, mW1p[t]); pack(mWo[t], DH_, DDEN, mWop[t]);
  }
  pack(oW0, DDEN, DH_, oW0p);   pack(oW1, DH_, DH_, oW1p);   pack(oWo, DH_, 1, oWop);

  pair_geom<<<(NPAIR + 255) / 256, 256, 0, stream>>>(cart, nl, sph, dist);

  const int mlpRows = MLP_WAVES * 16;  // rows per block
  mlp_wmma<<<(NPAIR + mlpRows - 1) / mlpRows, MLP_WAVES * 32, 0, stream>>>(
      input_embed, NPAIR, EMB_, nW0p, nb0, nW1p, nb1, nWop, nbo, DNEIGH, embN);
  mlp_wmma<<<(NPAIR + mlpRows - 1) / mlpRows, MLP_WAVES * 32, 0, stream>>>(
      input_embed, NPAIR, EMB_, cgW0p, cgb0, cgW1p, cgb1, cgWop, cgbo, MPL_ * NCGU_, embC);

  radial_k<<<(NPAIR + 255) / 256, 256, 0, stream>>>(embN, dist, radial);

  center0_k<<<NATOM, NC_, 0, stream>>>(ccoeff, radial, sph, embN, cenA, density);

  float* cold = cenA;
  float* cnew = cenB;
  for (int t = 0; t < MPL_; ++t) {
    mlp_wmma<<<(NATOM + mlpRows - 1) / mlpRows, MLP_WAVES * 32, 0, stream>>>(
        density, NATOM, DDEN, mW0p[t], mb0[t], mW1p[t], mb1[t], mWop[t], mbo[t], DDEN, coeff);
    interact_k<<<NATOM, NC_, 0, stream>>>(t, ccoeff, radial, sph, embC, coeff, nl,
                                          i1, i2, iadd, icg, ens_cg, cold, cnew, density);
    float* tmp = cold; cold = cnew; cnew = tmp;
  }

  mlp_wmma<<<(NATOM + mlpRows - 1) / mlpRows, MLP_WAVES * 32, 0, stream>>>(
      density, NATOM, DDEN, oW0p, ob0, oW1p, ob1, oWop, obo, 1, yout);
  reduce_k<<<1, 128, 0, stream>>>(yout, (float*)d_out);
}
